// Wp_grurefine_4303557231224
// MI455X (gfx1250) — matrix operand order repaired and checked by reading, not compiled, not run on hardware
//
#include <hip/hip_runtime.h>
#include <hip/hip_bf16.h>
#include <math.h>

typedef __attribute__((ext_vector_type(16))) __bf16 v16bf;
typedef __attribute__((ext_vector_type(8)))  float  v8f;

#define NEMBD     512
#define TSTEPS    8
#define HID       66
#define G3        198              // 3*HID
#define NT_G      13               // ceil(198/16) N-tiles
#define NPAD      208              // 13*16
#define KH_PAD    96               // hidden K padded (66 -> 96, 3 chunks of 32)
#define KX_PAD    32               // input K padded (16 -> 32, 1 chunk)
#define WAVES     4

#define H_STRIDE  96
#define LDS_H     (16*H_STRIDE)            // 1536 f
#define LDS_G     (16*NPAD)                // 3328 f
#define LDS_XIN   (16*KX_PAD)              // 512 f
#define LDS_X     32
#define PW_FLOATS (LDS_H + 2*LDS_G + LDS_XIN + LDS_X)  // 8736 floats per wave
// block-shared constants staged in LDS: b_ih[198] b_hh[198] W_out[132] b_out[2]
#define C_BIH     0
#define C_BHH     198
#define C_WOUT    396
#define C_BOUT    528
#define C_FLOATS  530

__device__ __forceinline__ void cvt8(const float* p, __bf16* dst) {
  float4 a = *(const float4*)p;
  float4 b = *(const float4*)(p + 4);
  dst[0] = (__bf16)a.x; dst[1] = (__bf16)a.y; dst[2] = (__bf16)a.z; dst[3] = (__bf16)a.w;
  dst[4] = (__bf16)b.x; dst[5] = (__bf16)b.y; dst[6] = (__bf16)b.z; dst[7] = (__bf16)b.w;
}
// A fragment from two runs of 8 contiguous fp32 (converted to bf16 in-register)
__device__ __forceinline__ v16bf cvt2x8(const float* p0, const float* p1) {
  union { v16bf v; __bf16 e[16]; } u;
  cvt8(p0, u.e); cvt8(p1, u.e + 8);
  return u.v;
}

__device__ __forceinline__ v8f wmma_bf16(v16bf a, v16bf b, v8f c) {
  return __builtin_amdgcn_wmma_f32_16x16x32_bf16(false, a, false, b, (short)0, c,
                                                 false, false);
}

// ---- pre-pass: convert/pad all GEMM weights to bf16 B-fragment-friendly rows ----
__global__ void cvt_weights(const float* __restrict__ W_ih,
                            const float* __restrict__ W_hh,
                            const float* __restrict__ W_head,
                            __bf16* __restrict__ Wih_p,     // [NPAD][KX_PAD]
                            __bf16* __restrict__ Whh_p,     // [NPAD][KH_PAD]
                            __bf16* __restrict__ Whead_p) { // [64][512]
  int i = blockIdx.x * blockDim.x + threadIdx.x;
  if (i < NPAD * KX_PAD) {
    int n = i >> 5, k = i & 31;
    Wih_p[i] = (__bf16)((n < G3 && k < 16) ? W_ih[n * 16 + k] : 0.0f);
  }
  if (i < NPAD * KH_PAD) {
    int n = i / KH_PAD, k = i - n * KH_PAD;
    Whh_p[i] = (__bf16)((n < G3 && k < HID) ? W_hh[n * HID + k] : 0.0f);
  }
  if (i < 64 * NEMBD) {
    Whead_p[i] = (__bf16)W_head[i];
  }
}

// ---- fused head-GEMM + 8-step GRU + waypoint output; one 16-row tile per wave ----
__global__ void __launch_bounds__(WAVES * 32)
wp_gru_fused(const float* __restrict__ x_last,    // [B,8,2]
             const float* __restrict__ cls_emb,   // [B,512]
             const float* __restrict__ tp,        // [B,2]
             const float* __restrict__ light,     // [B,1]
             const float* __restrict__ cmd,       // [B,1]
             const float* __restrict__ bbox,      // [B,1,8,10]
             const float* __restrict__ b_head,    // [64]
             const float* __restrict__ b_ih,      // [198]
             const float* __restrict__ b_hh,      // [198]
             const float* __restrict__ W_out,     // [2,66]
             const float* __restrict__ b_out,     // [2]
             const __bf16* __restrict__ Wih_p,    // [208][32]
             const __bf16* __restrict__ Whh_p,    // [208][96]
             const __bf16* __restrict__ Whead_p,  // [64][512]
             float* __restrict__ out)             // [B,8,2]
{
  extern __shared__ float smem[];
  const int lane = threadIdx.x & 31;
  const int wave = threadIdx.x >> 5;
  const int lml  = lane & 15;
  const int hi   = lane >> 4;           // half-wave select
  const int ksel = hi ? 16 : 0;         // A-frag K sub-offset (contiguous k = 16*hi + s)
  const int m0   = (blockIdx.x * WAVES + wave) * 16;

  float* hT  = smem + wave * PW_FLOATS; // [16][96] hidden (zero-padded K)
  float* giT = hT  + LDS_H;             // [16][208] gi staging
  float* ghT = giT + LDS_G;             // [16][208] gh staging
  float* xin = ghT + LDS_G;             // [16][32]  x_in (zero-padded K)
  float* xT  = xin + LDS_XIN;           // [16][2]   running waypoint
  float* cst = smem + WAVES * PW_FLOATS; // block-shared constants

  // stage small constants into LDS (block-cooperative)
  for (int i = threadIdx.x; i < C_FLOATS; i += WAVES * 32) {
    float v;
    if (i < C_BHH)            v = b_ih[i];
    else if (i < C_WOUT)      v = b_hh[i - C_BHH];
    else if (i < C_BOUT)      v = W_out[i - C_WOUT];
    else                      v = b_out[i - C_BOUT];
    cst[i] = v;
  }

  // ================= head GEMM: z0 = cls_emb @ W_head^T + b_head =================
  const float* Arow = cls_emb + (size_t)(m0 + lml) * NEMBD;
  #pragma unroll 1
  for (int n0 = 0; n0 < 4; ++n0) {
    v8f acc = {};
    const __bf16* Brow = Whead_p + (size_t)(n0 * 16 + lml) * NEMBD + hi * 16;
    #pragma unroll 4
    for (int kc = 0; kc < 16; ++kc) {
      const float* pa = Arow + kc * 32 + ksel;
      __builtin_prefetch(pa + 64, 0, 1);
      v16bf a = cvt2x8(pa, pa + 8);
      v16bf b = *(const v16bf*)(Brow + kc * 32);
      acc = wmma_bf16(a, b, acc);
    }
    int n = n0 * 16 + lml;
    float bh = b_head[n];
    #pragma unroll
    for (int r = 0; r < 8; ++r)
      hT[(hi * 8 + r) * H_STRIDE + n] = acc[r] + bh;   // C layout: M = hi*8+r, N = lml
  }
  if (lane < 16) {
    int b = m0 + lane;
    hT[lane * H_STRIDE + 64] = light[b];
    hT[lane * H_STRIDE + 65] = cmd[b];
    xT[lane * 2 + 0] = 0.0f;
    xT[lane * 2 + 1] = 0.0f;
  }
  for (int idx = lane; idx < 16 * (H_STRIDE - HID); idx += 32) {
    int m = idx / (H_STRIDE - HID);
    int c = HID + (idx - m * (H_STRIDE - HID));
    hT[m * H_STRIDE + c] = 0.0f;
  }
  __syncthreads();

  // ================= GRU: 8 steps =================
  for (int t = 0; t < TSTEPS; ++t) {
    // ---- gather x_in = [x, tp, x_last[t], bbox[t]] into padded LDS tile ----
    if (lane < 16) {
      int b = m0 + lane;
      float* row = xin + lane * KX_PAD;
      row[0] = xT[lane * 2 + 0];
      row[1] = xT[lane * 2 + 1];
      row[2] = tp[b * 2 + 0];
      row[3] = tp[b * 2 + 1];
      row[4] = x_last[((size_t)b * TSTEPS + t) * 2 + 0];
      row[5] = x_last[((size_t)b * TSTEPS + t) * 2 + 1];
      #pragma unroll
      for (int j = 0; j < 10; ++j)
        row[6 + j] = bbox[((size_t)b * TSTEPS + t) * 10 + j];
      #pragma unroll
      for (int j = 16; j < 32; ++j) row[j] = 0.0f;
    }
    __syncthreads();

    // ---- A fragments (bf16) ----
    const float* xrow = xin + lml * KX_PAD;
    v16bf a_x  = cvt2x8(xrow + ksel, xrow + ksel + 8);
    const float* hrow = hT + lml * H_STRIDE;
    v16bf a_h0 = cvt2x8(hrow +  0 + ksel, hrow +  8 + ksel);
    v16bf a_h1 = cvt2x8(hrow + 32 + ksel, hrow + 40 + ksel);
    v16bf a_h2 = cvt2x8(hrow + 64 + ksel, hrow + 72 + ksel);

    // ---- gi and gh over 13 N-tiles ----
    #pragma unroll 1
    for (int n0 = 0; n0 < NT_G; ++n0) {
      int n = n0 * 16 + lml;
      v8f acc = {};
      acc = wmma_bf16(a_x, *(const v16bf*)(Wih_p + n * KX_PAD + hi * 16), acc);
      #pragma unroll
      for (int r = 0; r < 8; ++r) giT[(hi * 8 + r) * NPAD + n] = acc[r];

      const __bf16* bw = Whh_p + n * KH_PAD + hi * 16;
      v8f acc2 = {};
      acc2 = wmma_bf16(a_h0, *(const v16bf*)(bw +  0), acc2);
      acc2 = wmma_bf16(a_h1, *(const v16bf*)(bw + 32), acc2);
      acc2 = wmma_bf16(a_h2, *(const v16bf*)(bw + 64), acc2);
      #pragma unroll
      for (int r = 0; r < 8; ++r) ghT[(hi * 8 + r) * NPAD + n] = acc2[r];
    }
    __syncthreads();

    // ---- gates (16*66 = 1056 elems, 33 per lane, no divergence) ----
    for (int idx = lane; idx < 16 * HID; idx += 32) {
      int m = idx / HID;
      int n = idx - m * HID;
      const float* gim = giT + m * NPAD;
      const float* ghm = ghT + m * NPAD;
      float gr  = gim[n]           + ghm[n]           + cst[C_BIH + n]           + cst[C_BHH + n];
      float gu  = gim[HID + n]     + ghm[HID + n]     + cst[C_BIH + HID + n]     + cst[C_BHH + HID + n];
      float gin = gim[2 * HID + n] + cst[C_BIH + 2 * HID + n];
      float ghn = ghm[2 * HID + n] + cst[C_BHH + 2 * HID + n];
      float r  = 1.0f / (1.0f + __expf(-gr));
      float u  = 1.0f / (1.0f + __expf(-gu));
      float nn = tanhf(gin + r * ghn);
      float hp = hT[m * H_STRIDE + n];
      hT[m * H_STRIDE + n] = (1.0f - u) * nn + u * hp;
    }
    __syncthreads();

    // ---- x = h_new @ W_out^T + b_out + x ; emit waypoint ----
    {
      int m = lane >> 1, c = lane & 1;
      const float* wrow = cst + C_WOUT + c * HID;
      const float* hm   = hT + m * H_STRIDE;
      float s = cst[C_BOUT + c] + xT[lane];
      #pragma unroll 2
      for (int j = 0; j < HID; ++j) s += hm[j] * wrow[j];
      xT[lane] = s;
      out[(((size_t)(m0 + m)) * TSTEPS + t) * 2 + c] = s - (c ? 0.0f : 1.3f);
    }
    __syncthreads();
  }
}

extern "C" void kernel_launch(void* const* d_in, const int* in_sizes, int n_in,
                              void* d_out, int out_size, void* d_ws, size_t ws_size,
                              hipStream_t stream) {
  const float* x_last  = (const float*)d_in[0];
  const float* cls_emb = (const float*)d_in[1];
  const float* tp      = (const float*)d_in[2];
  const float* light   = (const float*)d_in[3];
  const float* cmd     = (const float*)d_in[4];
  const float* bbox    = (const float*)d_in[5];
  const float* W_head  = (const float*)d_in[6];
  const float* b_head  = (const float*)d_in[7];
  const float* W_ih    = (const float*)d_in[8];
  const float* W_hh    = (const float*)d_in[9];
  const float* b_ih    = (const float*)d_in[10];
  const float* b_hh    = (const float*)d_in[11];
  const float* W_out   = (const float*)d_in[12];
  const float* b_out   = (const float*)d_in[13];
  float* out = (float*)d_out;

  __bf16* Wih_p   = (__bf16*)d_ws;                      // 208*32*2  = 13312 B
  __bf16* Whh_p   = (__bf16*)((char*)d_ws + 16384);     // 208*96*2  = 39936 B
  __bf16* Whead_p = (__bf16*)((char*)d_ws + 57344);     // 64*512*2  = 65536 B

  int Bn = in_sizes[1] / NEMBD;

  cvt_weights<<<(64 * NEMBD + 255) / 256, 256, 0, stream>>>(
      W_ih, W_hh, W_head, Wih_p, Whh_p, Whead_p);

  int blocks = Bn / (WAVES * 16);
  size_t lds_bytes = ((size_t)WAVES * PW_FLOATS + C_FLOATS) * sizeof(float);  // ~142 KB
  wp_gru_fused<<<blocks, WAVES * 32, lds_bytes, stream>>>(
      x_last, cls_emb, tp, light, cmd, bbox, b_head,
      b_ih, b_hh, W_out, b_out, Wih_p, Whh_p, Whead_p, out);
}
